// HV_CAB_encoder_84129819394212
// MI455X (gfx1250) — matrix operand order repaired and checked by reading, not compiled, not run on hardware
//
#include <hip/hip_runtime.h>
#include <hip/hip_bf16.h>

// CDNA5 / gfx1250 implementation of the HV-CAB channel-attention encoder.
// 5 kernels:
//   k_zero      : clear Gram/norm accumulators (atomically accumulated later)
//   k_qpath     : q = dw3x3(conv1x1(degra,q1w) * conv1x1(x,q2w))  -> HBM f16 (WMMA + pk_fma_f16)
//   k_kvpath    : kv = dw3x3(conv1x1(y,kvw)); write v f16; WMMA Gram G += q k^T; norm sums (atomics)
//   k_attn_proj : attn = softmax(G / (|q||k|) * temp); M[b] = proj_w @ blockdiag(attn)  (tiny)
//   k_out       : out[b,:,p] = M[b] @ v[b,:,p]   (per-batch 64x64 WMMA "conv1x1")
// gfx1250-specific paths: v_wmma_f32_16x16x32_f16, global_load_async_to_lds_b128
// (+ s_wait_asynccnt), global_prefetch_b8, packed f16 VALU for the depthwise taps.

typedef __attribute__((ext_vector_type(16))) _Float16 v16h;
typedef __attribute__((ext_vector_type(8)))  _Float16 v8h;
typedef __attribute__((ext_vector_type(8)))  float    v8f;

#define HEADS 4
#define CDIM  64
#define BATCH 8
#define HDIM  256
#define WDIM  256
#define HW    (HDIM * WDIM)

union UA  { v16h v; _Float16 e[16]; };
union UC  { v8f  v; float    e[8];  };
union UH8 { v8h  v; _Float16 e[8];  };

#define WMMA_F16(a, b, c) \
  __builtin_amdgcn_wmma_f32_16x16x32_f16(false, (a), false, (b), (short)0, (c), false, false)

// ---- async global->LDS (Tensor/async data path, ASYNCcnt tracked) -------
// Copies 16 bytes per enabled lane from global memory to LDS.
static __device__ __forceinline__ void async_copy_b128(unsigned lds_byte_off,
                                                       const void* gptr) {
  asm volatile("global_load_async_to_lds_b128 %0, %1, off"
               :
               : "v"(lds_byte_off), "v"((unsigned long long)gptr)
               : "memory");
}
static __device__ __forceinline__ void wait_async0() {
  asm volatile("s_wait_asynccnt 0x0" ::: "memory");
}
// low 32 bits of a generic pointer into LDS == LDS byte offset (aperture map)
static __device__ __forceinline__ unsigned lds_off32(const void* p) {
  return (unsigned)(unsigned long long)p;
}

// ---- fragment builders matching CDNA5 16-bit WMMA lane layouts ----------

// B fragment (KxN, 16-bit): lane n holds N=n, K = 0..7 & 16..23 (lanes 0-15) / 8..15 & 24..31 (lanes 16-31).
// Source: row-major [row=N][col=K] f16 buffer (weights, or k stored [ch][pix]).
static __device__ __forceinline__ v16h load_frag_B(const _Float16* buf, int stride,
                                                   int nbase, int kbase, int lane) {
  int n  = lane & 15;
  int kh = (lane >> 4) << 3;  // 0 or 8
  const _Float16* p = buf + (size_t)(nbase + n) * stride + kbase + kh;
  union { v16h v; v8h h[2]; } u;
  u.h[0] = *(const v8h*)p;
  u.h[1] = *(const v8h*)(p + 16);
  return u.v;
}

// A fragment (MxK, 16-bit) from LDS, source row-major [row=M][col=K]:
// lane m (0-15): K = 0..7 & 16..23 ; lane m+16: K = 8..15 & 24..31.
static __device__ __forceinline__ v16h load_frag_A_lds(const _Float16* buf, int stride,
                                                       int mbase, int kbase, int lane) {
  int m  = lane & 15;
  int kh = (lane >> 4) << 3;  // 0 or 8
  const _Float16* p = buf + (size_t)(mbase + m) * stride + kbase + kh;
  union { v16h v; v8h h[2]; } u;
  u.h[0] = *(const v8h*)p;         // K = kh+0..7
  u.h[1] = *(const v8h*)(p + 16);  // K = kh+16..23
  return u.v;
}

// A fragment where M = 16 consecutive pixels of one image row, K = channels,
// gathered from planar f32 NCHW global memory (channel-plane stride = HW),
// with zero-fill outside the image (conv halo).
static __device__ __forceinline__ v16h load_frag_A_g32(const float* plane, int kbase,
                                                       int row, int colbase, int lane) {
  int m   = lane & 15;
  int kh  = (lane >> 4) << 3;
  int col = colbase + m;
  UA a;
  if (row >= 0 && row < HDIM && col >= 0 && col < WDIM) {
    size_t pix = (size_t)row * WDIM + col;
#pragma unroll
    for (int i = 0; i < 8; ++i) {
      a.e[i]     = (_Float16)plane[(size_t)(kbase + kh + i) * HW + pix];
      a.e[i + 8] = (_Float16)plane[(size_t)(kbase + kh + 16 + i) * HW + pix];
    }
  } else {
#pragma unroll
    for (int i = 0; i < 16; ++i) a.e[i] = (_Float16)0.f;
  }
  return a.v;
}

// Same, from planar f16 NCHW (no bounds needed).
static __device__ __forceinline__ v16h load_frag_A_g16(const _Float16* plane, int kbase,
                                                       size_t pixbase, int lane) {
  int m  = lane & 15;
  int kh = (lane >> 4) << 3;
  size_t pix = pixbase + m;
  UA a;
#pragma unroll
  for (int i = 0; i < 8; ++i) {
    a.e[i]     = plane[(size_t)(kbase + kh + i) * HW + pix];
    a.e[i + 8] = plane[(size_t)(kbase + kh + 16 + i) * HW + pix];
  }
  return a.v;
}

// ------------------------------------------------------------------------
__global__ void k_zero(float* __restrict__ p, int n) {
  int i = blockIdx.x * blockDim.x + threadIdx.x;
  if (i < n) p[i] = 0.f;
}

// ---- q path: tile 8 rows x 64 cols, halo 10 x 80 (5 chunks of 16) -------
// LDS: qpre[800][72] f16 + w1[64][72] + w2[64][72] + dwq[9][64]  = 134784 B
__global__ void k_qpath(const float* __restrict__ degra, const float* __restrict__ x,
                        const float* __restrict__ q1w, const float* __restrict__ q2w,
                        const float* __restrict__ qdw, _Float16* __restrict__ qout) {
  extern __shared__ __align__(16) char smem[];
  _Float16* qpre = (_Float16*)smem;        // [800][72]
  _Float16* w1   = qpre + 800 * 72;        // [64][72]
  _Float16* w2   = w1 + 64 * 72;           // [64][72]
  _Float16* dwq  = w2 + 64 * 72;           // [9][64]

  const int tid = threadIdx.x, lane = tid & 31, wave = tid >> 5;
  const int b  = blockIdx.z;
  const int r0 = blockIdx.y * 8;
  const int c0 = blockIdx.x * 64;

  for (int i = tid; i < 64 * 64; i += 256) {
    w1[(i >> 6) * 72 + (i & 63)] = (_Float16)q1w[i];
    w2[(i >> 6) * 72 + (i & 63)] = (_Float16)q2w[i];
  }
  for (int i = tid; i < 64 * 9; i += 256) {
    int ch = i / 9, tap = i % 9;
    dwq[tap * 64 + ch] = (_Float16)qdw[i];
  }
  __syncthreads();

  const float* dplane = degra + (size_t)b * CDIM * HW;
  const float* xplane = x     + (size_t)b * CDIM * HW;

  // fused 1x1 convs + elementwise product -> qpre (f16, pixel-major)
  for (int t = wave; t < 50; t += 8) {           // 10 halo rows x 5 col-chunks
    int hr = t / 5, chk = t % 5;
    int row     = r0 - 1 + hr;
    int colbase = c0 - 1 + chk * 16;
    int pixbase = hr * 80 + chk * 16;

    // prefetch next tile's rows (global_prefetch_b8; speculative, OOB-safe)
    int tn = t + 8;
    if (tn < 50) {
      int rown = r0 - 1 + tn / 5;
      int coln = c0 - 1 + (tn % 5) * 16 + (lane & 15);
      if (rown >= 0 && rown < HDIM && coln >= 0 && coln < WDIM) {
        size_t pixn = (size_t)((lane >> 4) << 3) * HW + (size_t)rown * WDIM + coln;
        __builtin_prefetch(dplane + pixn, 0, 1);
        __builtin_prefetch(xplane + pixn, 0, 1);
      }
    }

    v16h aD0 = load_frag_A_g32(dplane, 0,  row, colbase, lane);
    v16h aD1 = load_frag_A_g32(dplane, 32, row, colbase, lane);
    v16h aX0 = load_frag_A_g32(xplane, 0,  row, colbase, lane);
    v16h aX1 = load_frag_A_g32(xplane, 32, row, colbase, lane);
#pragma unroll
    for (int nt = 0; nt < 4; ++nt) {
      v8f c1 = {}; v8f c2 = {};
      c1 = WMMA_F16(aD0, load_frag_B(w1, 72, nt * 16, 0,  lane), c1);
      c1 = WMMA_F16(aD1, load_frag_B(w1, 72, nt * 16, 32, lane), c1);
      c2 = WMMA_F16(aX0, load_frag_B(w2, 72, nt * 16, 0,  lane), c2);
      c2 = WMMA_F16(aX1, load_frag_B(w2, 72, nt * 16, 32, lane), c2);
      UC p1, p2; p1.v = c1; p2.v = c2;
      int och  = nt * 16 + (lane & 15);
      int mtop = (lane >> 4) << 3;
#pragma unroll
      for (int i = 0; i < 8; ++i)
        qpre[(pixbase + mtop + i) * 72 + och] = (_Float16)(p1.e[i] * p2.e[i]);
    }
  }
  __syncthreads();

  // depthwise 3x3 (v_pk_fma_f16) + store q as planar f16 NCHW
  _Float16* qplane = qout + (size_t)b * CDIM * HW;
  for (int pp = tid; pp < 512; pp += 256) {
    int orow = pp >> 6, ocol = pp & 63;
    size_t gpix = (size_t)(r0 + orow) * WDIM + (c0 + ocol);
#pragma unroll
    for (int cb = 0; cb < 8; ++cb) {
      v8h acc = {};
#pragma unroll
      for (int dr = 0; dr < 3; ++dr)
#pragma unroll
        for (int dc = 0; dc < 3; ++dc) {
          const v8h s  = *(const v8h*)&qpre[((orow + dr) * 80 + (ocol + dc)) * 72 + cb * 8];
          const v8h wv = *(const v8h*)&dwq[(dr * 3 + dc) * 64 + cb * 8];
          acc += s * wv;
        }
      UH8 u; u.v = acc;
#pragma unroll
      for (int i = 0; i < 8; ++i)
        qplane[(size_t)(cb * 8 + i) * HW + gpix] = u.e[i];
    }
  }
}

// ---- kv path: tile 8 x 32, halo 10 x 48; also Gram + norms --------------
// LDS: kvpre[480][136] + wkv[128][72] + dwv[9][128] + kbuf[64][264] + qbuf[64][264] = 218880 B
__global__ void k_kvpath(const float* __restrict__ y, const float* __restrict__ kvw,
                         const float* __restrict__ kvdw, const _Float16* __restrict__ qin,
                         _Float16* __restrict__ vout, float* __restrict__ G,
                         float* __restrict__ nq, float* __restrict__ nk) {
  extern __shared__ __align__(16) char smem[];
  _Float16* kvpre = (_Float16*)smem;       // [480][136]
  _Float16* wkv   = kvpre + 480 * 136;     // [128][72]
  _Float16* dwv   = wkv + 128 * 72;        // [9][128]
  _Float16* kbuf  = dwv + 9 * 128;         // [64][264]  k, transposed [ch][pix]
  _Float16* qbuf  = kbuf + 64 * 264;       // [64][264]  q tile, [ch][pix]

  const int tid = threadIdx.x, lane = tid & 31, wave = tid >> 5;
  const int b  = blockIdx.z;
  const int r0 = blockIdx.y * 8;
  const int c0 = blockIdx.x * 32;

  // stage q tile (produced by k_qpath, f16) via async global->LDS copies:
  // 64 ch x 8 rows x (32 px = 4 chunks of 16B) = 2048 lane-chunks, 8 rounds.
  {
    const _Float16* qplane = qin + (size_t)b * CDIM * HW;
    unsigned qoff = lds_off32(qbuf);
    for (int idx = tid; idx < 2048; idx += 256) {
      int ch    = idx >> 5;          // 0..63
      int trow  = (idx & 31) >> 2;   // 0..7
      int tcol8 = (idx & 3) * 8;     // halves 0,8,16,24
      const _Float16* src =
          qplane + (size_t)ch * HW + (size_t)(r0 + trow) * WDIM + c0 + tcol8;
      async_copy_b128(qoff + (unsigned)(ch * 528 + trow * 64 + tcol8 * 2), src);
    }
  }

  for (int i = tid; i < 128 * 64; i += 256)
    wkv[(i >> 6) * 72 + (i & 63)] = (_Float16)kvw[i];
  for (int i = tid; i < 128 * 9; i += 256) {
    int ch = i / 9, tap = i % 9;
    dwv[tap * 128 + ch] = (_Float16)kvdw[i];
  }
  wait_async0();
  __syncthreads();

  const float* yplane = y + (size_t)b * CDIM * HW;
  for (int t = wave; t < 30; t += 8) {          // 10 halo rows x 3 col-chunks
    int hr = t / 3, chk = t % 3;
    int row     = r0 - 1 + hr;
    int colbase = c0 - 1 + chk * 16;
    int pixbase = hr * 48 + chk * 16;

    int tn = t + 8;
    if (tn < 30) {
      int rown = r0 - 1 + tn / 3;
      int coln = c0 - 1 + (tn % 3) * 16 + (lane & 15);
      if (rown >= 0 && rown < HDIM && coln >= 0 && coln < WDIM)
        __builtin_prefetch(
            yplane + (size_t)((lane >> 4) << 3) * HW + (size_t)rown * WDIM + coln, 0, 1);
    }

    v16h a0 = load_frag_A_g32(yplane, 0,  row, colbase, lane);
    v16h a1 = load_frag_A_g32(yplane, 32, row, colbase, lane);
#pragma unroll
    for (int nt = 0; nt < 8; ++nt) {            // 128 output channels
      v8f c = {};
      c = WMMA_F16(a0, load_frag_B(wkv, 72, nt * 16, 0,  lane), c);
      c = WMMA_F16(a1, load_frag_B(wkv, 72, nt * 16, 32, lane), c);
      UC u; u.v = c;
      int och  = nt * 16 + (lane & 15);
      int mtop = (lane >> 4) << 3;
#pragma unroll
      for (int i = 0; i < 8; ++i)
        kvpre[(pixbase + mtop + i) * 136 + och] = (_Float16)u.e[i];
    }
  }
  __syncthreads();

  // depthwise 3x3: k -> kbuf (transposed), v -> global f16 NCHW
  {
    int pix  = tid;                  // 256 threads == 256 tile pixels
    int orow = pix >> 5, ocol = pix & 31;
    size_t gpix = (size_t)(r0 + orow) * WDIM + (c0 + ocol);
    _Float16* vplane = vout + (size_t)b * CDIM * HW;
#pragma unroll
    for (int cb = 0; cb < 16; ++cb) {
      v8h acc = {};
#pragma unroll
      for (int dr = 0; dr < 3; ++dr)
#pragma unroll
        for (int dc = 0; dc < 3; ++dc) {
          const v8h s  = *(const v8h*)&kvpre[((orow + dr) * 48 + (ocol + dc)) * 136 + cb * 8];
          const v8h wv = *(const v8h*)&dwv[(dr * 3 + dc) * 128 + cb * 8];
          acc += s * wv;
        }
      UH8 u; u.v = acc;
      if (cb < 8) {
#pragma unroll
        for (int i = 0; i < 8; ++i) kbuf[(cb * 8 + i) * 264 + pix] = u.e[i];
      } else {
#pragma unroll
        for (int i = 0; i < 8; ++i)
          vplane[(size_t)((cb - 8) * 8 + i) * HW + gpix] = u.e[i];
      }
    }
  }
  __syncthreads();

  // Gram G[b,h] += q_tile k_tile^T via WMMA: M=16 q-ch, N=16 k-ch, K=pixels
  {
    int hd   = wave & 3;
    int pseg = (wave >> 2) * 128;
    v8f g = {};
#pragma unroll
    for (int kk = 0; kk < 4; ++kk) {
      int pbase = pseg + kk * 32;
      v16h a  = load_frag_A_lds(qbuf, 264, hd * 16, pbase, lane);
      v16h bb = load_frag_B   (kbuf, 264, hd * 16, pbase, lane);
      g = WMMA_F16(a, bb, g);
    }
    UC u; u.v = g;
    float* gbase = G + ((size_t)b * HEADS + hd) * 256;
    int d = lane & 15, mtop = (lane >> 4) << 3;
#pragma unroll
    for (int i = 0; i < 8; ++i)
      atomicAdd(&gbase[(mtop + i) * 16 + d], u.e[i]);
  }

  // squared-norm partial sums
  {
    int ch = tid & 63, seg = tid >> 6;
    float sq = 0.f, sk = 0.f;
    for (int p = seg * 64; p < seg * 64 + 64; ++p) {
      float qv = (float)qbuf[ch * 264 + p];
      float kv = (float)kbuf[ch * 264 + p];
      sq += qv * qv; sk += kv * kv;
    }
    atomicAdd(&nq[b * 64 + ch], sq);
    atomicAdd(&nk[b * 64 + ch], sk);
  }
}

// ---- softmax + fold proj: M[b] = proj_w @ blockdiag(attn[b]) ------------
__global__ void k_attn_proj(const float* __restrict__ G, const float* __restrict__ nq,
                            const float* __restrict__ nk, const float* __restrict__ temp,
                            const float* __restrict__ projw, _Float16* __restrict__ Mh) {
  __shared__ float attnS[HEADS][16][16];
  const int b = blockIdx.x, tid = threadIdx.x;
  if (tid < 64) {
    int h = tid >> 4, c = tid & 15;
    float qn  = fmaxf(sqrtf(nq[b * 64 + h * 16 + c]), 1e-12f);
    float tpr = temp[h];
    float s[16];
    float mx = -1e30f;
    for (int d = 0; d < 16; ++d) {
      float kn = fmaxf(sqrtf(nk[b * 64 + h * 16 + d]), 1e-12f);
      float v  = G[((size_t)b * HEADS + h) * 256 + c * 16 + d] / (qn * kn) * tpr;
      s[d] = v; mx = fmaxf(mx, v);
    }
    float sum = 0.f;
    for (int d = 0; d < 16; ++d) { s[d] = __expf(s[d] - mx); sum += s[d]; }
    float inv = 1.f / sum;
    for (int d = 0; d < 16; ++d) attnS[h][c][d] = s[d] * inv;
  }
  __syncthreads();
  int o = tid & 63, h = tid >> 6;
  for (int j = 0; j < 16; ++j) {
    float acc = 0.f;
    for (int c = 0; c < 16; ++c)
      acc += projw[o * 64 + h * 16 + c] * attnS[h][c][j];
    Mh[(size_t)b * 4096 + o * 64 + h * 16 + j] = (_Float16)acc;
  }
}

// ---- out = M[b] @ v : per-batch 64x64 "conv1x1" over pixels -------------
// LDS: wm[64][72] f16 + stage[8][64][16] f32 = 41984 B
__global__ void k_out(const _Float16* __restrict__ vin, const _Float16* __restrict__ Mh,
                      float* __restrict__ out) {
  extern __shared__ __align__(16) char smem[];
  _Float16* wm = (_Float16*)smem;                 // [64][72]
  float* stage = (float*)(smem + 64 * 72 * 2);    // [8 waves][64 ch][16 px]
  const int tid = threadIdx.x, lane = tid & 31, wave = tid >> 5;
  const int b = blockIdx.y;

  // stage M[b] (f16, 8KB) via async global->LDS: 64 rows x 8 chunks of 16B
  {
    const _Float16* src0 = Mh + (size_t)b * 4096;
    unsigned woff = lds_off32(wm);
    for (int idx = tid; idx < 512; idx += 256) {
      int o = idx >> 3, j = idx & 7;   // row, 16B chunk in row
      async_copy_b128(woff + (unsigned)(o * 144 + j * 16), src0 + o * 64 + j * 8);
    }
  }
  wait_async0();
  __syncthreads();

  size_t pbase = ((size_t)blockIdx.x * 8 + wave) * 16;   // 16 pixels per wave
  const _Float16* vplane = vin + (size_t)b * CDIM * HW;
  v16h a0 = load_frag_A_g16(vplane, 0,  pbase, lane);
  v16h a1 = load_frag_A_g16(vplane, 32, pbase, lane);
  float* stw = stage + wave * 1024;
#pragma unroll
  for (int nt = 0; nt < 4; ++nt) {
    v8f c = {};
    c = WMMA_F16(a0, load_frag_B(wm, 72, nt * 16, 0,  lane), c);
    c = WMMA_F16(a1, load_frag_B(wm, 72, nt * 16, 32, lane), c);
    UC u; u.v = c;
    int ch   = nt * 16 + (lane & 15);
    int mtop = (lane >> 4) << 3;
#pragma unroll
    for (int i = 0; i < 8; ++i) stw[ch * 16 + mtop + i] = u.e[i];
  }
  __syncthreads();   // order LDS transpose staging before writeback

  float* oplane = out + (size_t)b * CDIM * HW;
#pragma unroll
  for (int cc = 0; cc < 2; ++cc) {
    int ch = lane * 2 + cc;
#pragma unroll
    for (int j = 0; j < 4; ++j)
      *(float4*)&oplane[(size_t)ch * HW + pbase + j * 4] =
          *(const float4*)&stw[ch * 16 + j * 4];
  }
}

// ------------------------------------------------------------------------
extern "C" void kernel_launch(void* const* d_in, const int* in_sizes, int n_in,
                              void* d_out, int out_size, void* d_ws, size_t ws_size,
                              hipStream_t stream) {
  (void)in_sizes; (void)n_in; (void)out_size; (void)ws_size;
  const float* degra = (const float*)d_in[0];
  const float* x     = (const float*)d_in[1];
  const float* y     = (const float*)d_in[2];
  const float* q1w   = (const float*)d_in[3];
  const float* q2w   = (const float*)d_in[4];
  const float* qdw   = (const float*)d_in[5];
  const float* kvw   = (const float*)d_in[6];
  const float* kvdw  = (const float*)d_in[7];
  const float* projw = (const float*)d_in[8];
  const float* temp  = (const float*)d_in[9];
  float* out = (float*)d_out;

  // workspace carve-out (needs ~134.3 MB)
  char* ws = (char*)d_ws;
  _Float16* qbuf = (_Float16*)ws;                                   // B*C*HW f16
  _Float16* vbuf = (_Float16*)(ws + (size_t)BATCH * CDIM * HW * 2); // B*C*HW f16
  float* G  = (float*)(ws + (size_t)BATCH * CDIM * HW * 4);         // B*4*16*16
  float* nq = G + BATCH * HEADS * 256;                              // B*64
  float* nk = nq + BATCH * CDIM;                                    // B*64
  _Float16* Mh = (_Float16*)(nk + BATCH * CDIM);                    // B*64*64 f16

  int nacc = BATCH * HEADS * 256 + 2 * BATCH * CDIM;                // 9216 floats
  k_zero<<<dim3((nacc + 255) / 256), dim3(256), 0, stream>>>(G, nacc);
  k_qpath<<<dim3(WDIM / 64, HDIM / 8, BATCH), dim3(256), 134784, stream>>>(
      degra, x, q1w, q2w, qdw, qbuf);
  k_kvpath<<<dim3(WDIM / 32, HDIM / 8, BATCH), dim3(256), 218880, stream>>>(
      y, kvw, kvdw, qbuf, vbuf, G, nq, nk);
  k_attn_proj<<<dim3(BATCH), dim3(256), 0, stream>>>(G, nq, nk, temp, projw, Mh);
  k_out<<<dim3(HW / 128, BATCH), dim3(256), 41984, stream>>>(vbuf, Mh, out);
}
